// NexusNet_12197707120678
// MI455X (gfx1250) — matrix operand order repaired and checked by reading, not compiled, not run on hardware
//
#include <hip/hip_runtime.h>

// ---------------------------------------------------------------------------
// NuGraph-style nexus GNN block for MI455X (gfx1250, wave32, WMMA).
// All class-linear layers run on v_wmma_f32_16x16x32_bf16 (bf16 in, f32 acc).
// Scatter-sums (plane->nexus, edge->plane) use global_atomic_add_f32.
// Inter-layer fragment re-layout is staged through per-wave LDS tiles.
// ---------------------------------------------------------------------------

#define C_  5
#define NF_ 64
#define SF_ 16
#define NP_ 50000
#define NX_ 30000
#define E_  200000

typedef float  v8f   __attribute__((ext_vector_type(8)));
typedef __bf16 v16bf __attribute__((ext_vector_type(16)));

__device__ __forceinline__ __bf16 to_bf16(float f) { return (__bf16)f; }

__device__ __forceinline__ v8f vzero8() {
  v8f v;
#pragma unroll
  for (int i = 0; i < 8; ++i) v[i] = 0.f;
  return v;
}

__device__ __forceinline__ v8f wmma_bf16(v16bf a, v16bf b, v8f c) {
  // D = A(16x32 bf16) * B(32x16 bf16) + C(16x16 f32)
  return __builtin_amdgcn_wmma_f32_16x16x32_bf16(false, a, false, b,
                                                 (short)0, c, false, false);
}

// A fragment: 16x32 tile from row-major f32 source. Lane L holds row (L&15),
// K = (L>>4)*8 + {0..7} in elems [0..7] and +16 in elems [8..15].
__device__ __forceinline__ v16bf load_a_frag(const float* base, int stride,
                                             int lane, int k0, int kvalid) {
  int m  = lane & 15;
  int kb = (lane >> 4) * 8;
  const float* row = base + (size_t)m * stride;
  v16bf a;
#pragma unroll
  for (int i = 0; i < 8; ++i) {
    int k1 = k0 + kb + i;
    int k2 = k0 + 16 + kb + i;
    a[i]     = to_bf16(k1 < kvalid ? row[k1] : 0.f);
    a[8 + i] = to_bf16(k2 < kvalid ? row[k2] : 0.f);
  }
  return a;
}

// B fragment: B[k][n] = W[n][k], W row-major [16 out][kin]. Lane L holds
// col (L&15), K = (L>>4)*8 + {0..7} in elems [0..7] and +16 in elems [8..15].
__device__ __forceinline__ v16bf load_b_frag(const float* W, int kin,
                                             int lane, int k0, int kvalid) {
  int n  = lane & 15;
  int kb = (lane >> 4) * 8;
  const float* row = W + (size_t)n * kin;
  v16bf b;
#pragma unroll
  for (int j = 0; j < 16; ++j) {
    int k = k0 + kb + (j & 7) + ((j >> 3) << 4);
    b[j] = to_bf16(k < kvalid ? row[k] : 0.f);
  }
  return b;
}

// ---------------------------------------------------------------------------
__global__ void zero_kernel(float* __restrict__ p, long long n) {
  long long i = (long long)blockIdx.x * blockDim.x + threadIdx.x;
  if (i < n) p[i] = 0.f;
}

// Plane -> nexus scatter-add (segment_sum). 80 threads per edge, float4 each.
// ups layout: [NX][C][3*NF], plane p occupies feature slice [p*NF, (p+1)*NF).
__global__ void nexus_up_kernel(const float* __restrict__ x,
                                const int* __restrict__ src,
                                const int* __restrict__ dst,
                                float* __restrict__ ups, int plane) {
  long long t = (long long)blockIdx.x * blockDim.x + threadIdx.x;
  if (t >= (long long)E_ * 80) return;
  int e = (int)(t / 80);
  int j = (int)(t % 80);
  int c = j >> 4;
  int f = (j & 15) << 2;
  int s = src[e], d = dst[e];
  const float4 v = *(const float4*)(x + (size_t)s * (C_ * NF_) + c * NF_ + f);
  float* dp = ups + (size_t)d * (C_ * 3 * NF_) + c * (3 * NF_) + plane * NF_ + f;
  atomicAdd(dp + 0, v.x);
  atomicAdd(dp + 1, v.y);
  atomicAdd(dp + 2, v.z);
  atomicAdd(dp + 3, v.w);
}

// Nexus MLP: n = tanh(W2 @ tanh(W1 @ ups + b1) + b2). One wave per (tile,class).
__global__ void __launch_bounds__(256)
nexus_net_kernel(const float* __restrict__ ups,
                 const float* __restrict__ W1, const float* __restrict__ b1,
                 const float* __restrict__ W2, const float* __restrict__ b2,
                 float* __restrict__ nbuf) {
  __shared__ float t_tile[8][16 * 17];
  int wave = threadIdx.x >> 5, lane = threadIdx.x & 31;
  int gw = blockIdx.x * 8 + wave;
  const int TILES = NX_ / 16;
  if (gw >= TILES * C_) return;
  int tile = gw / C_, c = gw % C_;

  const float* abase = ups + (size_t)tile * 16 * (C_ * 3 * NF_) + c * (3 * NF_);
  v8f acc = vzero8();
#pragma unroll
  for (int kc = 0; kc < 6; ++kc) {  // K = 192
    v16bf a = load_a_frag(abase, C_ * 3 * NF_, lane, kc * 32, 3 * NF_);
    v16bf b = load_b_frag(W1 + (size_t)c * SF_ * (3 * NF_), 3 * NF_, lane, kc * 32, 3 * NF_);
    acc = wmma_bf16(a, b, acc);
  }
  int n = lane & 15, hi = lane >> 4;
  float* tt = t_tile[wave];
#pragma unroll
  for (int r = 0; r < 8; ++r)
    tt[(r + 8 * hi) * 17 + n] = tanhf(acc[r] + b1[c * SF_ + n]);

  // layer 2: 16x16, K padded to 32 with zeros
  v16bf afrag2 = load_a_frag(tt, 17, lane, 0, SF_);
  v16bf bfrag2 = load_b_frag(W2 + (size_t)c * SF_ * SF_, SF_, lane, 0, SF_);
  v8f acc2 = wmma_bf16(afrag2, bfrag2, vzero8());
#pragma unroll
  for (int r = 0; r < 8; ++r) {
    int mrow = tile * 16 + r + 8 * hi;
    nbuf[(size_t)mrow * (C_ * SF_) + c * SF_ + n] = tanhf(acc2[r] + b2[c * SF_ + n]);
  }
}

__device__ __forceinline__ float m_feat(const float* __restrict__ x,
                                        const float* __restrict__ nbuf,
                                        int s, int d, int c, int k) {
  if (k < NF_) return x[(size_t)s * (C_ * NF_) + c * NF_ + k];
  return nbuf[(size_t)d * (C_ * SF_) + c * SF_ + (k - NF_)];
}

// Edge attention + message scatter. One wave per tile of 16 edges.
__global__ void __launch_bounds__(256)
edge_kernel(const float* __restrict__ x, const int* __restrict__ src,
            const int* __restrict__ dst, const float* __restrict__ nbuf,
            const float* __restrict__ eW1, const float* __restrict__ eb1,
            const float* __restrict__ eW2, const float* __restrict__ eb2,
            float* __restrict__ numb, float* __restrict__ cntb) {
  __shared__ float h_tile[8][16 * 17];
  __shared__ float lg[8][16 * 8];
  __shared__ int   sd[8][32];
  int wave = threadIdx.x >> 5, lane = threadIdx.x & 31;
  int tile = blockIdx.x * 8 + wave;
  const int TILES = E_ / 16;
  if (tile >= TILES) return;

  float* ht  = h_tile[wave];
  float* lgt = lg[wave];
  int*   sdt = sd[wave];

  int m = lane & 15, hi = lane >> 4;
  int e_lane = tile * 16 + m;
  int s_lane = src[e_lane];
  int d_lane = dst[e_lane];
  if (lane < 16) { sdt[m] = s_lane; sdt[16 + m] = d_lane; }

  for (int c = 0; c < C_; ++c) {
    v8f acc = vzero8();
#pragma unroll
    for (int kc = 0; kc < 3; ++kc) {  // K = 80, padded to 96
      int k0 = kc * 32, kb = hi * 8;
      v16bf a;
#pragma unroll
      for (int i = 0; i < 8; ++i) {
        int k1 = k0 + kb + i;
        int k2 = k0 + 16 + kb + i;
        a[i]     = to_bf16(k1 < (NF_ + SF_) ? m_feat(x, nbuf, s_lane, d_lane, c, k1) : 0.f);
        a[8 + i] = to_bf16(k2 < (NF_ + SF_) ? m_feat(x, nbuf, s_lane, d_lane, c, k2) : 0.f);
      }
      v16bf b = load_b_frag(eW1 + (size_t)c * SF_ * (NF_ + SF_), NF_ + SF_,
                            lane, k0, NF_ + SF_);
      acc = wmma_bf16(a, b, acc);
    }
    int n = lane & 15;
#pragma unroll
    for (int r = 0; r < 8; ++r)
      ht[(r + 8 * hi) * 17 + n] = tanhf(acc[r] + eb1[c * SF_ + n]);
    // second edge layer: per-row dot with eW2[c] -> logit (lanes 0..15)
    if (lane < 16) {
      float lgv = eb2[c];
#pragma unroll
      for (int f = 0; f < SF_; ++f) lgv += ht[lane * 17 + f] * eW2[c * SF_ + f];
      lgt[lane * 8 + c] = lgv;
    }
  }

  // softmax over classes (lanes 0..15, one row each) + degree count
  if (lane < 16) {
    float mx = -1e30f;
#pragma unroll
    for (int c = 0; c < C_; ++c) mx = fmaxf(mx, lgt[lane * 8 + c]);
    float wv[C_]; float sum = 0.f;
#pragma unroll
    for (int c = 0; c < C_; ++c) { wv[c] = __expf(lgt[lane * 8 + c] - mx); sum += wv[c]; }
    float inv = 1.f / sum;
#pragma unroll
    for (int c = 0; c < C_; ++c) lgt[lane * 8 + c] = wv[c] * inv;
    atomicAdd(&cntb[s_lane], 1.0f);
  }

  // msg = w * n_j scattered to num[src]: 16 rows * 5 classes * 16 feats = 1280
#pragma unroll
  for (int iter = 0; iter < 40; ++iter) {
    int id  = iter * 32 + lane;
    int mm  = id / (C_ * SF_);
    int rem = id % (C_ * SF_);
    int c   = rem / SF_;
    int f   = rem % SF_;
    int s = sdt[mm], d = sdt[16 + mm];
    float val = lgt[mm * 8 + c] * nbuf[(size_t)d * (C_ * SF_) + c * SF_ + f];
    atomicAdd(&numb[(size_t)s * (C_ * SF_) + c * SF_ + f], val);
  }
}

__device__ __forceinline__ float u_feat(const float* __restrict__ x,
                                        const float* __restrict__ numb,
                                        float invc, int row, int c, int k) {
  if (k < NF_) return x[(size_t)row * (C_ * NF_) + c * NF_ + k];
  return numb[(size_t)row * (C_ * SF_) + c * SF_ + (k - NF_)] * invc;
}

// Node MLP: out = tanh(W2 @ tanh(W1 @ [x|aggr] + b1) + b2). Wave per (tile,class).
__global__ void __launch_bounds__(256)
node_kernel(const float* __restrict__ x, const float* __restrict__ numb,
            const float* __restrict__ cntb,
            const float* __restrict__ nW1, const float* __restrict__ nb1,
            const float* __restrict__ nW2, const float* __restrict__ nb2,
            float* __restrict__ outp) {
  __shared__ float t_tile[8][16 * 65];
  int wave = threadIdx.x >> 5, lane = threadIdx.x & 31;
  int gw = blockIdx.x * 8 + wave;
  const int TILES = NP_ / 16;
  if (gw >= TILES * C_) return;
  int tile = gw / C_, c = gw % C_;
  int m = lane & 15, hi = lane >> 4;
  int row = tile * 16 + m;
  float invc = 1.f / fmaxf(cntb[row], 1.f);

  v8f acc[4];
#pragma unroll
  for (int nt = 0; nt < 4; ++nt) acc[nt] = vzero8();
#pragma unroll
  for (int kc = 0; kc < 3; ++kc) {  // K = 80, padded to 96
    int k0 = kc * 32, kb = hi * 8;
    v16bf a;
#pragma unroll
    for (int i = 0; i < 8; ++i) {
      int k1 = k0 + kb + i;
      int k2 = k0 + 16 + kb + i;
      a[i]     = to_bf16(k1 < (NF_ + SF_) ? u_feat(x, numb, invc, row, c, k1) : 0.f);
      a[8 + i] = to_bf16(k2 < (NF_ + SF_) ? u_feat(x, numb, invc, row, c, k2) : 0.f);
    }
#pragma unroll
    for (int nt = 0; nt < 4; ++nt) {
      v16bf b = load_b_frag(nW1 + (size_t)c * NF_ * (NF_ + SF_) + (size_t)nt * 16 * (NF_ + SF_),
                            NF_ + SF_, lane, k0, NF_ + SF_);
      acc[nt] = wmma_bf16(a, b, acc[nt]);
    }
  }
  float* tt = t_tile[wave];
  int n = lane & 15;
#pragma unroll
  for (int nt = 0; nt < 4; ++nt)
#pragma unroll
    for (int r = 0; r < 8; ++r)
      tt[(r + 8 * hi) * 65 + nt * 16 + n] = tanhf(acc[nt][r] + nb1[c * NF_ + nt * 16 + n]);

  v8f acc2[4];
#pragma unroll
  for (int nt = 0; nt < 4; ++nt) acc2[nt] = vzero8();
#pragma unroll
  for (int kc = 0; kc < 2; ++kc) {  // K = 64
    v16bf a = load_a_frag(tt, 65, lane, kc * 32, NF_);
#pragma unroll
    for (int nt = 0; nt < 4; ++nt) {
      v16bf b = load_b_frag(nW2 + (size_t)c * NF_ * NF_ + (size_t)nt * 16 * NF_,
                            NF_, lane, kc * 32, NF_);
      acc2[nt] = wmma_bf16(a, b, acc2[nt]);
    }
  }
#pragma unroll
  for (int nt = 0; nt < 4; ++nt)
#pragma unroll
    for (int r = 0; r < 8; ++r) {
      int mrow = tile * 16 + r + 8 * hi;
      outp[(size_t)mrow * (C_ * NF_) + c * NF_ + nt * 16 + n] =
          tanhf(acc2[nt][r] + nb2[c * NF_ + nt * 16 + n]);
    }
}

// ---------------------------------------------------------------------------
extern "C" void kernel_launch(void* const* d_in, const int* in_sizes, int n_in,
                              void* d_out, int out_size, void* d_ws, size_t ws_size,
                              hipStream_t stream) {
  (void)in_sizes; (void)n_in; (void)out_size; (void)ws_size;

  const float* xs[3]   = {(const float*)d_in[0], (const float*)d_in[3], (const float*)d_in[6]};
  const int*   srcs[3] = {(const int*)d_in[1],   (const int*)d_in[4],   (const int*)d_in[7]};
  const int*   dsts[3] = {(const int*)d_in[2],   (const int*)d_in[5],   (const int*)d_in[8]};
  // d_in[9] = initial nexus features (only its shape is used by the reference)

  // params leaves in jax tree order: 'down' (3 x {edge:[(W,b),(W,b)], node:[(W,b),(W,b)]}),
  // then 'nexus_net' ([(W,b),(W,b)])
  const int base = 10;
  const float *eW1[3], *eb1[3], *eW2[3], *eb2[3], *nW1[3], *nb1[3], *nW2[3], *nb2[3];
  for (int p = 0; p < 3; ++p) {
    const int o = base + p * 8;
    eW1[p] = (const float*)d_in[o + 0];
    eb1[p] = (const float*)d_in[o + 1];
    eW2[p] = (const float*)d_in[o + 2];
    eb2[p] = (const float*)d_in[o + 3];
    nW1[p] = (const float*)d_in[o + 4];
    nb1[p] = (const float*)d_in[o + 5];
    nW2[p] = (const float*)d_in[o + 6];
    nb2[p] = (const float*)d_in[o + 7];
  }
  const float* xW1 = (const float*)d_in[34];
  const float* xb1 = (const float*)d_in[35];
  const float* xW2 = (const float*)d_in[36];
  const float* xb2 = (const float*)d_in[37];

  // Workspace layout (f32): ups [NX][C][192] | nbuf [NX][C][16] | num [NP][C][16] | cnt [NP]
  float* ws   = (float*)d_ws;
  float* ups  = ws;
  float* nbuf = ups  + (size_t)NX_ * (C_ * 3 * NF_);
  float* numb = nbuf + (size_t)NX_ * (C_ * SF_);
  float* cntb = numb + (size_t)NP_ * (C_ * SF_);

  float* out = (float*)d_out;

  // 1) zero nexus accumulator, scatter-add all planes, run nexus MLP
  {
    long long nz = (long long)NX_ * (C_ * 3 * NF_);
    zero_kernel<<<(unsigned)((nz + 255) / 256), 256, 0, stream>>>(ups, nz);
  }
  {
    long long nt = (long long)E_ * 80;
    unsigned blocks = (unsigned)((nt + 255) / 256);
    for (int p = 0; p < 3; ++p)
      nexus_up_kernel<<<blocks, 256, 0, stream>>>(xs[p], srcs[p], dsts[p], ups, p);
  }
  {
    int waves = (NX_ / 16) * C_;
    nexus_net_kernel<<<(waves + 7) / 8, 256, 0, stream>>>(ups, xW1, xb1, xW2, xb2, nbuf);
  }

  // 2) per plane: zero num/cnt, edge attention + scatter, node MLP
  for (int p = 0; p < 3; ++p) {
    long long nz = (long long)NP_ * (C_ * SF_) + NP_;  // num + cnt contiguous
    zero_kernel<<<(unsigned)((nz + 255) / 256), 256, 0, stream>>>(numb, nz);

    int etiles = E_ / 16;
    edge_kernel<<<(etiles + 7) / 8, 256, 0, stream>>>(
        xs[p], srcs[p], dsts[p], nbuf, eW1[p], eb1[p], eW2[p], eb2[p], numb, cntb);

    int nwaves = (NP_ / 16) * C_;
    node_kernel<<<(nwaves + 7) / 8, 256, 0, stream>>>(
        xs[p], numb, cntb, nW1[p], nb1[p], nW2[p], nb2[p],
        out + (size_t)p * NP_ * (C_ * NF_));
  }
}
